// Chess_Model_Actor_41154376630834
// MI455X (gfx1250) — matrix operand order repaired and checked by reading, not compiled, not run on hardware
//
#include <hip/hip_runtime.h>

// ---------------- model dims ----------------
#define DMODEL 768
#define SLEN   65
#define BATCH  4
#define NTOK   (BATCH*SLEN)      // 260
#define HEADS  4
#define DKV    64
#define INNERD 256
#define DFF    2048
#define NEXP   32
#define CAPAC  64
#define NACT   1968
#define FLATK  (DMODEL*SLEN)     // 49920
#define NLAY   4

// output layout (floats), concat in reference return order
#define O_OUT1   0
#define O_OUT3   (BATCH*NACT)                       // 7872
#define O_LAST   (O_OUT3 + NTOK*14)                 // 11512
#define O_VAL    (O_LAST + BATCH*DMODEL)            // 14584
#define O_ROUTER (O_VAL + BATCH)                    // 14588 (2 * 260 * 32 follows)

// gemm flags
#define FLAG_RELU  1
#define FLAG_ACCUM 2
#define FLAG_F32   4
#define FLAG_BF16  8

typedef __attribute__((ext_vector_type(16))) __bf16 bf16x16;
typedef __attribute__((ext_vector_type(8)))  __bf16 bf16x8;
typedef __attribute__((ext_vector_type(8)))  float  f32x8;

// ---------------- embedding ----------------
__global__ void k_embed(const int* in1, const int* in2, const float* pose,
                        const float* piece, float* x) {
  int idx = blockIdx.x * blockDim.x + threadIdx.x;
  if (idx >= NTOK * DMODEL) return;
  int t = idx / DMODEL, d = idx % DMODEL;
  x[idx] = pose[in1[t] * DMODEL + d] + piece[in2[t] * DMODEL + d];
}

// ---------------- T5 relative position bias ----------------
__global__ void k_bias(const float* rel_bias, float* bias) {
  int idx = blockIdx.x * blockDim.x + threadIdx.x;
  if (idx >= HEADS * SLEN * SLEN) return;
  int h = idx / (SLEN * SLEN);
  int r = idx % (SLEN * SLEN);
  int i = r / SLEN, j = r % SLEN;
  int rel = j - i;                       // memory - context
  int ret = rel > 0 ? 16 : 0;
  int n = rel < 0 ? -rel : rel;
  int b;
  if (n < 8) {
    b = n;
  } else {
    float nf = (float)n;
    int large = 8 + (int)(logf(nf / 8.0f) / logf(16.0f) * 8.0f);
    b = large < 15 ? large : 15;
  }
  bias[idx] = rel_bias[(ret + b) * HEADS + h];
}

// ---------------- RMS norm (writes f32 + bf16 copies) ----------------
__global__ void k_rmsnorm(const float* x, const float* w, float* hf, __bf16* hb) {
  int t = blockIdx.x;
  const float* xr = x + (long)t * DMODEL;
  __shared__ float red[256];
  float ss = 0.f;
  for (int d = threadIdx.x; d < DMODEL; d += 256) { float v = xr[d]; ss += v * v; }
  red[threadIdx.x] = ss; __syncthreads();
  for (int s = 128; s > 0; s >>= 1) {
    if (threadIdx.x < s) red[threadIdx.x] += red[threadIdx.x + s];
    __syncthreads();
  }
  float r = rsqrtf(red[0] / (float)DMODEL + 1e-6f);
  for (int d = threadIdx.x; d < DMODEL; d += 256) {
    float v = w[d] * xr[d] * r;
    hf[(long)t * DMODEL + d] = v;
    hb[(long)t * DMODEL + d] = (__bf16)v;
  }
}

// ---------------- wave-level WMMA 32x16 tile (2 accumulators share B) ----------
// A: bf16 row-major [.,lda], optionally row-gathered via rowIdx; out-of-range
//    rows/cols are CLAMPED (their products land only in unstored D rows/cols).
// W: fp32 row-major [K,N], N compile-time -> immediate-offset column loads.
// No divergent branches inside: EXEC stays all-1s for every WMMA.
template <int N>
__device__ __forceinline__ void wmma_loop2(const __bf16* A, long lda, const int* rowIdx,
                                           int m0, int M, const float* W, int K,
                                           int n0, f32x8& acc0, f32x8& acc1) {
  int lane = threadIdx.x & 31;
  int mloc = lane & 15;
  int s    = lane >> 4;
  int r0 = m0 + mloc;
  int r1 = m0 + 16 + mloc;
  int c0i = r0 < M ? r0 : M - 1;                 // clamp, store-guarded later
  int c1i = r1 < M ? r1 : M - 1;
  long arow0 = rowIdx ? (long)rowIdx[c0i] : (long)c0i;
  long arow1 = rowIdx ? (long)rowIdx[c1i] : (long)c1i;
  int ncol = n0 + (lane & 15);
  if (ncol >= N) ncol = N - 1;                   // clamp, store-guarded later
  const __bf16* ap0 = A + arow0 * lda + 8 * s;   // 16B-aligned chunks
  const __bf16* ap1 = A + arow1 * lda + 8 * s;
  const float*  wp  = W + (long)(8 * s) * N + ncol;
  for (int k0 = 0; k0 < K; k0 += 32) {
    bf16x8 a0lo = *(const bf16x8*)ap0;
    bf16x8 a0hi = *(const bf16x8*)(ap0 + 16);
    bf16x8 a1lo = *(const bf16x8*)ap1;
    bf16x8 a1hi = *(const bf16x8*)(ap1 + 16);
    bf16x16 b;
#pragma unroll
    for (int j = 0; j < 16; ++j) b[j] = (__bf16)wp[(j + (j >> 3) * 8) * N];  // imm-offset loads
    bf16x16 a0 = __builtin_shufflevector(a0lo, a0hi,
                 0, 1, 2, 3, 4, 5, 6, 7, 8, 9, 10, 11, 12, 13, 14, 15);
    bf16x16 a1 = __builtin_shufflevector(a1lo, a1hi,
                 0, 1, 2, 3, 4, 5, 6, 7, 8, 9, 10, 11, 12, 13, 14, 15);
    acc0 = __builtin_amdgcn_wmma_f32_16x16x32_bf16(false, a0, false, b, (short)0,
                                                   acc0, false, false);
    acc1 = __builtin_amdgcn_wmma_f32_16x16x32_bf16(false, a1, false, b, (short)0,
                                                   acc1, false, false);
    ap0 += 32; ap1 += 32; wp += 32 * N;          // running pointers only
  }
}

__device__ __forceinline__ void wmma_init(f32x8& acc, const float* C, long ldc,
                                          int m0, int M, int n0, int N, bool accum) {
  int lane = threadIdx.x & 31;
  int s = lane >> 4;
  int ncol = n0 + (lane & 15);
#pragma unroll
  for (int r = 0; r < 8; ++r) {
    float v = 0.f;
    if (accum) {
      int mrow = m0 + r + 8 * s;
      if (mrow < M && ncol < N) v = C[(long)mrow * ldc + ncol];
    }
    acc[r] = v;
  }
}

__device__ __forceinline__ void wmma_store(f32x8& acc, float* C, long ldc,
                                           __bf16* Cb, long ldcb,
                                           int m0, int M, int n0, int N, int flags,
                                           const int* rowIdxC, const float* rowScale) {
  int lane = threadIdx.x & 31;
  int s = lane >> 4;
  int ncol = n0 + (lane & 15);
  if (ncol >= N) return;
#pragma unroll
  for (int r = 0; r < 8; ++r) {
    int mrow = m0 + r + 8 * s;
    if (mrow >= M) continue;
    float v = acc[r];
    if (flags & FLAG_RELU) v = v > 0.f ? v : 0.f;
    if (rowIdxC) {                 // scatter: unique writer per element, no atomics
      int tok = rowIdxC[mrow];
      C[(long)tok * ldc + ncol] += rowScale[tok] * v;
    } else {
      if (flags & FLAG_F32)  C [(long)mrow * ldc  + ncol] = v;
      if (flags & FLAG_BF16) Cb[(long)mrow * ldcb + ncol] = (__bf16)v;
    }
  }
}

// ---------------- generic GEMM: block = 4 waves, each wave = 32x16 tile ----------
template <int N>
__global__ void k_gemm(const __bf16* A, long lda, const float* W,
                       float* C, long ldc, __bf16* Cb, long ldcb,
                       int M, int K, int flags) {
  int wave = threadIdx.x >> 5;
  int n0 = blockIdx.x * 64 + wave * 16;
  int m0 = blockIdx.y * 32;
  if (n0 >= N || m0 >= M) return;            // wave-uniform: EXEC stays full
  bool accum = (flags & FLAG_ACCUM) != 0;
  f32x8 acc0, acc1;
  wmma_init(acc0, C, ldc, m0,      M, n0, N, accum);
  wmma_init(acc1, C, ldc, m0 + 16, M, n0, N, accum);
  wmma_loop2<N>(A, lda, nullptr, m0, M, W, K, n0, acc0, acc1);
  wmma_store(acc0, C, ldc, Cb, ldcb, m0,      M, n0, N, flags, nullptr, nullptr);
  wmma_store(acc1, C, ldc, Cb, ldcb, m0 + 16, M, n0, N, flags, nullptr, nullptr);
}

// ---------------- attention (small: per (b,h) in LDS) ----------------
__global__ void k_attn(const float* q, const float* k, const float* v,
                       const float* bias, __bf16* ctxb) {
  int b = blockIdx.x / HEADS;
  int h = blockIdx.x % HEADS;
  __shared__ float ks[SLEN * DKV];
  __shared__ float vs[SLEN * DKV];
  for (int idx = threadIdx.x; idx < SLEN * DKV; idx += blockDim.x) {
    int s = idx / DKV, d = idx % DKV;
    long off = (long)(b * SLEN + s) * INNERD + h * DKV + d;
    ks[idx] = k[off];
    vs[idx] = v[off];
  }
  __syncthreads();
  int i = threadIdx.x;
  if (i >= SLEN) return;
  float qr[DKV];
  for (int d = 0; d < DKV; ++d) qr[d] = q[(long)(b * SLEN + i) * INNERD + h * DKV + d];
  float sc[SLEN];
  float mx = -1e30f;
  for (int jj = 0; jj < SLEN; ++jj) {
    float acc = bias[(h * SLEN + i) * SLEN + jj];   // no 1/sqrt(d) scaling (T5)
    for (int d = 0; d < DKV; ++d) acc += qr[d] * ks[jj * DKV + d];
    sc[jj] = acc;
    mx = fmaxf(mx, acc);
  }
  float sum = 0.f;
  for (int jj = 0; jj < SLEN; ++jj) { float e = __expf(sc[jj] - mx); sc[jj] = e; sum += e; }
  float inv = 1.0f / sum;
  for (int d = 0; d < DKV; ++d) {
    float acc = 0.f;
    for (int jj = 0; jj < SLEN; ++jj) acc += sc[jj] * vs[jj * DKV + d];
    ctxb[(long)(b * SLEN + i) * INNERD + h * DKV + d] = (__bf16)(acc * inv);
  }
}

// ---------------- deterministic routing (argmax + capacity cumsum) ----------------
__global__ void k_route(const float* logits, int* cnt, int* toklist, int* kept,
                        float* maxp) {
  for (int e = 0; e < NEXP; ++e) cnt[e] = 0;
  for (int b = 0; b < BATCH; ++b) {
    int cbe[NEXP];
    for (int e = 0; e < NEXP; ++e) cbe[e] = 0;
    for (int s = 0; s < SLEN; ++s) {
      int t = b * SLEN + s;
      const float* lr = logits + (long)t * NEXP;
      float lmax = lr[0]; int eidx = 0;
      for (int e = 1; e < NEXP; ++e) if (lr[e] > lmax) { lmax = lr[e]; eidx = e; }
      float sum = 0.f;
      for (int e = 0; e < NEXP; ++e) sum += __expf(lr[e] - lmax);
      maxp[t] = 1.0f / sum;                 // softmax max prob
      cbe[eidx]++;
      if (cbe[eidx] <= CAPAC) {
        kept[t] = 1;
        toklist[eidx * 256 + cnt[eidx]] = t;
        cnt[eidx]++;
      } else {
        kept[t] = 0;
      }
    }
  }
}

// ---------------- per-expert FF, GEMM1: h @ wi_exp[e], relu, bf16 out ----------------
__global__ void k_expert_ffn1(const __bf16* hb, const float* wi_exp_j,
                              const int* cnt, const int* toklist, __bf16* ehid) {
  int wave = threadIdx.x >> 5;
  int e  = blockIdx.y >> 3;
  int m0 = (blockIdx.y & 7) * 32;
  int M  = cnt[e];
  int n0 = blockIdx.x * 64 + wave * 16;
  if (m0 >= M || n0 >= DFF) return;
  f32x8 acc0, acc1;
#pragma unroll
  for (int r = 0; r < 8; ++r) { acc0[r] = 0.f; acc1[r] = 0.f; }
  wmma_loop2<DFF>(hb, DMODEL, toklist + e * 256, m0, M,
                  wi_exp_j + (long)e * DMODEL * DFF, DMODEL, n0, acc0, acc1);
  __bf16* outp = ehid + (long)e * 256 * DFF;
  wmma_store(acc0, nullptr, 0, outp, DFF, m0,      M, n0, DFF,
             FLAG_RELU | FLAG_BF16, nullptr, nullptr);
  wmma_store(acc1, nullptr, 0, outp, DFF, m0 + 16, M, n0, DFF,
             FLAG_RELU | FLAG_BF16, nullptr, nullptr);
}

// ---------------- per-expert FF, GEMM2: hid @ wo_exp[e], scatter maxp*out ----------------
__global__ void k_expert_ffn2(const __bf16* ehid, const float* wo_exp_j,
                              const int* cnt, const int* toklist,
                              const float* maxp, float* x) {
  int wave = threadIdx.x >> 5;
  int e  = blockIdx.y >> 3;
  int m0 = (blockIdx.y & 7) * 32;
  int M  = cnt[e];
  int n0 = blockIdx.x * 64 + wave * 16;
  if (m0 >= M || n0 >= DMODEL) return;
  f32x8 acc0, acc1;
#pragma unroll
  for (int r = 0; r < 8; ++r) { acc0[r] = 0.f; acc1[r] = 0.f; }
  wmma_loop2<DMODEL>(ehid + (long)e * 256 * DFF, DFF, nullptr, m0, M,
                     wo_exp_j + (long)e * DFF * DMODEL, DFF, n0, acc0, acc1);
  wmma_store(acc0, x, DMODEL, nullptr, 0, m0,      M, n0, DMODEL, 0,
             toklist + e * 256, maxp);
  wmma_store(acc1, x, DMODEL, nullptr, 0, m0 + 16, M, n0, DMODEL, 0,
             toklist + e * 256, maxp);
}

// ---------------- dropped-token passthrough: x += maxp * h ----------------
__global__ void k_passthru(const int* kept, const float* maxp, const float* h, float* x) {
  int idx = blockIdx.x * blockDim.x + threadIdx.x;
  if (idx >= NTOK * DMODEL) return;
  int t = idx / DMODEL;
  if (!kept[t]) x[idx] += maxp[t] * h[idx];
}

// ---------------- heads epilogue: b1+mask, b3, last-token copy ----------------
__global__ void k_epilogue(float* out, const float* mask, const float* b1,
                           const float* b3, const float* enc) {
  int idx = blockIdx.x * blockDim.x + threadIdx.x;
  if (idx < BATCH * NACT) {
    float v = out[O_OUT1 + idx] + b1[idx % NACT];
    if (mask[idx] < -1.0f) v = -1e30f;
    out[O_OUT1 + idx] = v;
  } else if (idx < BATCH * NACT + NTOK * 14) {
    int i2 = idx - BATCH * NACT;
    out[O_OUT3 + i2] += b3[i2 % 14];
  } else if (idx < BATCH * NACT + NTOK * 14 + BATCH * DMODEL) {
    int i2 = idx - BATCH * NACT - NTOK * 14;
    int b = i2 / DMODEL, d = i2 % DMODEL;
    out[O_LAST + i2] = enc[(long)(b * SLEN + SLEN - 1) * DMODEL + d];
  }
}

// ---------------- value head: tanh(flat @ Wr + br) ----------------
__global__ void k_value(const float* enc, const float* Wr, const float* br, float* out) {
  int b = blockIdx.x;
  __shared__ float red[256];
  float ss = 0.f;
  const float* row = enc + (long)b * FLATK;
  for (int kk = threadIdx.x; kk < FLATK; kk += 256) ss += row[kk] * Wr[kk];
  red[threadIdx.x] = ss; __syncthreads();
  for (int s = 128; s > 0; s >>= 1) {
    if (threadIdx.x < s) red[threadIdx.x] += red[threadIdx.x + s];
    __syncthreads();
  }
  if (threadIdx.x == 0) out[O_VAL + b] = tanhf(red[0] + br[0]);
}

// ---------------- host launch ----------------
extern "C" void kernel_launch(void* const* d_in, const int* in_sizes, int n_in,
                              void* d_out, int out_size, void* d_ws, size_t ws_size,
                              hipStream_t stream) {
  const int*   input1   = (const int*)  d_in[0];
  const int*   input2   = (const int*)  d_in[1];
  const float* mask     = (const float*)d_in[2];
  const float* pos_emb  = (const float*)d_in[3];
  const float* piece    = (const float*)d_in[4];
  const float* ln1      = (const float*)d_in[5];
  const float* ln2      = (const float*)d_in[6];
  const float* wq       = (const float*)d_in[7];
  const float* wk       = (const float*)d_in[8];
  const float* wv       = (const float*)d_in[9];
  const float* wo       = (const float*)d_in[10];
  const float* rel_bias = (const float*)d_in[11];
  const float* wi_dense = (const float*)d_in[12];
  const float* wo_dense = (const float*)d_in[13];
  const float* router_w = (const float*)d_in[14];
  const float* wi_exp   = (const float*)d_in[15];
  const float* wo_exp   = (const float*)d_in[16];
  const float* final_ln = (const float*)d_in[17];
  const float* W1       = (const float*)d_in[18];
  const float* b1       = (const float*)d_in[19];
  const float* W3       = (const float*)d_in[20];
  const float* b3       = (const float*)d_in[21];
  const float* Wr       = (const float*)d_in[22];
  const float* br       = (const float*)d_in[23];
  float* out = (float*)d_out;

  // workspace carve-out (256B aligned), ~39 MB total
  char* base = (char*)d_ws;
  size_t off = 0;
  auto alloc = [&](size_t bytes) -> void* {
    off = (off + 255) & ~(size_t)255;
    void* p = base + off;
    off += bytes;
    return p;
  };
  float*  x     = (float*) alloc((size_t)NTOK * DMODEL * 4);
  float*  h     = (float*) alloc((size_t)NTOK * DMODEL * 4);
  __bf16* hb    = (__bf16*)alloc((size_t)NTOK * DMODEL * 2);
  float*  qbuf  = (float*) alloc((size_t)NTOK * INNERD * 4);
  float*  kbuf  = (float*) alloc((size_t)NTOK * INNERD * 4);
  float*  vbuf  = (float*) alloc((size_t)NTOK * INNERD * 4);
  __bf16* ctxb  = (__bf16*)alloc((size_t)NTOK * INNERD * 2);
  float*  bias  = (float*) alloc((size_t)HEADS * SLEN * SLEN * 4);
  __bf16* hidb  = (__bf16*)alloc((size_t)NTOK * DFF * 2);
  float*  enc   = (float*) alloc((size_t)NTOK * DMODEL * 4);
  __bf16* encb  = (__bf16*)alloc((size_t)NTOK * DMODEL * 2);
  int*    kept  = (int*)   alloc((size_t)NTOK * 4);
  float*  maxp  = (float*) alloc((size_t)NTOK * 4);
  int*    cnt   = (int*)   alloc((size_t)NEXP * 4);
  int*    tokl  = (int*)   alloc((size_t)NEXP * 256 * 4);
  __bf16* ehid  = (__bf16*)alloc((size_t)NEXP * 256 * DFF * 2);

  k_embed<<<(NTOK * DMODEL + 255) / 256, 256, 0, stream>>>(input1, input2, pos_emb,
                                                           piece, x);
  k_bias<<<(HEADS * SLEN * SLEN + 255) / 256, 256, 0, stream>>>(rel_bias, bias);

  dim3 blk(128);
  int my = (NTOK + 31) / 32;   // 9 m-tiles of 32 rows
  for (int i = 0; i < NLAY; ++i) {
    int j = i / 2;
    // ---- attention ----
    k_rmsnorm<<<NTOK, 256, 0, stream>>>(x, ln1 + (size_t)i * DMODEL, h, hb);
    dim3 gqkv((INNERD + 63) / 64, my);
    k_gemm<INNERD><<<gqkv, blk, 0, stream>>>(hb, DMODEL,
        wq + (size_t)i * DMODEL * INNERD, qbuf, INNERD, nullptr, 0,
        NTOK, DMODEL, FLAG_F32);
    k_gemm<INNERD><<<gqkv, blk, 0, stream>>>(hb, DMODEL,
        wk + (size_t)i * DMODEL * INNERD, kbuf, INNERD, nullptr, 0,
        NTOK, DMODEL, FLAG_F32);
    k_gemm<INNERD><<<gqkv, blk, 0, stream>>>(hb, DMODEL,
        wv + (size_t)i * DMODEL * INNERD, vbuf, INNERD, nullptr, 0,
        NTOK, DMODEL, FLAG_F32);
    k_attn<<<BATCH * HEADS, 128, 0, stream>>>(qbuf, kbuf, vbuf, bias, ctxb);
    dim3 gwo((DMODEL + 63) / 64, my);
    k_gemm<DMODEL><<<gwo, blk, 0, stream>>>(ctxb, INNERD,
        wo + (size_t)i * INNERD * DMODEL, x, DMODEL, nullptr, 0,
        NTOK, INNERD, FLAG_ACCUM | FLAG_F32);
    // ---- feed-forward ----
    k_rmsnorm<<<NTOK, 256, 0, stream>>>(x, ln2 + (size_t)i * DMODEL, h, hb);
    if ((i & 1) == 0) {   // dense FF
      dim3 gwi((DFF + 63) / 64, my);
      k_gemm<DFF><<<gwi, blk, 0, stream>>>(hb, DMODEL,
          wi_dense + (size_t)j * DMODEL * DFF, nullptr, 0, hidb, DFF,
          NTOK, DMODEL, FLAG_RELU | FLAG_BF16);
      k_gemm<DMODEL><<<gwo, blk, 0, stream>>>(hidb, DFF,
          wo_dense + (size_t)j * DFF * DMODEL, x, DMODEL, nullptr, 0,
          NTOK, DFF, FLAG_ACCUM | FLAG_F32);
    } else {              // Switch MoE FF (routed, not dense-all-experts)
      float* logits = out + O_ROUTER + (size_t)j * NTOK * NEXP;
      dim3 grt(1, my);                         // N=32 fits one 64-wide tile row
      k_gemm<NEXP><<<grt, blk, 0, stream>>>(hb, DMODEL,
          router_w + (size_t)j * DMODEL * NEXP, logits, NEXP, nullptr, 0,
          NTOK, DMODEL, FLAG_F32);
      k_route<<<1, 1, 0, stream>>>(logits, cnt, tokl, kept, maxp);
      dim3 ge1((DFF + 63) / 64, NEXP * 8);     // 8 m-tiles of 32 rows per expert
      k_expert_ffn1<<<ge1, blk, 0, stream>>>(hb,
          wi_exp + (size_t)j * NEXP * DMODEL * DFF, cnt, tokl, ehid);
      dim3 ge2((DMODEL + 63) / 64, NEXP * 8);
      k_expert_ffn2<<<ge2, blk, 0, stream>>>(ehid,
          wo_exp + (size_t)j * NEXP * DFF * DMODEL, cnt, tokl, maxp, x);
      k_passthru<<<(NTOK * DMODEL + 255) / 256, 256, 0, stream>>>(kept, maxp, h, x);
    }
  }

  // ---- final norm + heads ----
  k_rmsnorm<<<NTOK, 256, 0, stream>>>(x, final_ln, enc, encb);
  dim3 g1((NACT + 63) / 64, 1);               // M=4, K=49920 (flat view of encb)
  k_gemm<NACT><<<g1, blk, 0, stream>>>(encb, FLATK, W1, out + O_OUT1, NACT,
                                       nullptr, 0, BATCH, FLATK, FLAG_F32);
  dim3 g3(1, my);                             // N=14 ragged, guarded
  k_gemm<14><<<g3, blk, 0, stream>>>(encb, DMODEL, W3, out + O_OUT3, 14,
                                     nullptr, 0, NTOK, DMODEL, FLAG_F32);
  int epi_n = BATCH * NACT + NTOK * 14 + BATCH * DMODEL;
  k_epilogue<<<(epi_n + 255) / 256, 256, 0, stream>>>(out, mask, b1, b3, enc);
  k_value<<<BATCH, 256, 0, stream>>>(enc, Wr, br, out);
}
